// JambaBlock_37297495998519
// MI455X (gfx1250) — matrix operand order repaired and checked by reading, not compiled, not run on hardware
//
#include <hip/hip_runtime.h>
#include <hip/hip_bf16.h>

// ---------------------------------------------------------------------------
// CDNA5 (gfx1250) wave32 WMMA implementation of the sparse-MLA Jamba block.
// Matrix math via v_wmma_f32_16x16x32_bf16. LDS staging (GEMM tiles and
// attention K/V tiles) via the Tensor Data Mover with double buffering in
// the GEMM (tensor_load_to_lds + s_wait_tensorcnt).
// ---------------------------------------------------------------------------

typedef __attribute__((ext_vector_type(16))) __bf16 v16bf;
typedef __attribute__((ext_vector_type(8)))  __bf16 v8bf;
typedef __attribute__((ext_vector_type(8)))  float  v8f;
typedef __attribute__((ext_vector_type(4)))  unsigned int u32x4;
typedef __attribute__((ext_vector_type(8)))  int  i32x8;
typedef __attribute__((ext_vector_type(4)))  int  i32x4;

union BF16x16 { v16bf v; v8bf h[2]; };

#define D_MODEL 2048
#define T_SEQ   2048
#define BATCH   2
#define NH      16
#define NKV     4
#define HD      128
#define LATENT  512
#define WINDOW  256
#define GLOBAL_EVERY 64

#if __has_builtin(__builtin_amdgcn_tensor_load_to_lds) && \
    __has_builtin(__builtin_amdgcn_s_wait_tensorcnt)
#define USE_TDM 1
#else
#define USE_TDM 0
#endif

#if USE_TDM
// TDM 2D tile load: rows x tile_w bf16 elements from row-major global memory
// (row stride = row_stride elements) into LDS at lds_off. Hardware padding
// appends pad_amount=4 DWORDs (16B) after every 2^(1+ivl) DWORDs, letting the
// LDS image carry a +8-element row pad (ivl=3 for 32-wide rows -> stride 40,
// ivl=5 for 128-wide rows -> stride 136).
__device__ __forceinline__ void tdm_load_tile(const __bf16* gsrc,
                                              unsigned int lds_off,
                                              int rows, int tile_w,
                                              int row_stride, int ivl) {
  unsigned long long ga = (unsigned long long)(size_t)gsrc;
  u32x4 g0;
  g0[0] = 1u;                                   // count=1 valid descriptor
  g0[1] = lds_off;                              // lds_addr (bytes)
  g0[2] = (unsigned int)ga;                     // global_addr[31:0]
  g0[3] = (unsigned int)((ga >> 32) & 0x1FFFFFFu) | (2u << 30); // addr hi, type=2
  i32x8 g1;
  // data_size=1 (2B), pad_enable, pad_interval=ivl, pad_amount=4 DWORDs
  g1[0] = (1 << 16) | (1 << 20) | (ivl << 22) | (3 << 25);
  g1[1] = (int)((row_stride & 0xFFFF) << 16);   // tensor_dim0 lo @ bits 63:48
  g1[2] = (int)(((row_stride >> 16) & 0xFFFF) | ((rows & 0xFFFF) << 16));
  g1[3] = (int)(((rows >> 16) & 0xFFFF) | (tile_w << 16));   // dim1 hi, tile_dim0
  g1[4] = rows & 0xFFFF;                        // tile_dim1=rows, tile_dim2=0
  g1[5] = row_stride;                           // tensor_dim0_stride[31:0]
  g1[6] = 0;
  g1[7] = 0;
  i32x4 gz;  gz[0] = 0; gz[1] = 0; gz[2] = 0; gz[3] = 0;
#if __clang_major__ >= 23
  i32x8 gz8; for (int i = 0; i < 8; ++i) gz8[i] = 0;
  __builtin_amdgcn_tensor_load_to_lds(g0, g1, gz, gz, gz8, 0);
#else
  __builtin_amdgcn_tensor_load_to_lds(g0, g1, gz, gz, 0);
#endif
}
__device__ __forceinline__ unsigned int lds_byte_off(const void* p) {
  // LDS aperture: flat address low 32 bits are the LDS byte offset
  return (unsigned int)(unsigned long long)(size_t)p;
}
#endif

// ---------------------------------------------------------------------------
// 1) proj = t_emb @ W_ada^T + b_ada   (2 x 4096, tiny GEMV)
// ---------------------------------------------------------------------------
__global__ void ada_proj_kernel(const float* __restrict__ t_emb,
                                const float* __restrict__ W_ada,
                                const float* __restrict__ b_ada,
                                float* __restrict__ proj) {
  int idx = blockIdx.x * blockDim.x + threadIdx.x;   // 0..8191
  if (idx >= BATCH * 2 * D_MODEL) return;
  int b = idx >> 12;            // / 4096
  int o = idx & 4095;
  const float* te = t_emb + (size_t)b * D_MODEL;
  const float* wr = W_ada + (size_t)o * D_MODEL;
  float acc = b_ada[o];
  for (int c = 0; c < D_MODEL; ++c) acc += te[c] * wr[c];
  proj[idx] = acc;
}

// ---------------------------------------------------------------------------
// 2) fp32 -> bf16 weight casts
// ---------------------------------------------------------------------------
__global__ void f32_to_bf16_kernel(const float* __restrict__ src,
                                   __bf16* __restrict__ dst, int n) {
  int i = blockIdx.x * blockDim.x + threadIdx.x;
  if (i < n) dst[i] = (__bf16)src[i];
}

// ---------------------------------------------------------------------------
// 3) LayerNorm + adaLN modulation, output bf16 h
// ---------------------------------------------------------------------------
__global__ __launch_bounds__(256)
void ln_mod_kernel(const float* __restrict__ x,
                   const float* __restrict__ proj,
                   __bf16* __restrict__ h) {
  int token = blockIdx.x;                       // 0 .. B*T-1
  int b = token >> 11;
  const float* xr = x + (size_t)token * D_MODEL;
  __shared__ float rs[256], rs2[256];
  float s = 0.f, s2 = 0.f;
  for (int c = threadIdx.x; c < D_MODEL; c += 256) {
    float v = xr[c]; s += v; s2 += v * v;
  }
  rs[threadIdx.x] = s; rs2[threadIdx.x] = s2;
  __syncthreads();
  for (int off = 128; off > 0; off >>= 1) {
    if (threadIdx.x < off) {
      rs[threadIdx.x]  += rs[threadIdx.x + off];
      rs2[threadIdx.x] += rs2[threadIdx.x + off];
    }
    __syncthreads();
  }
  float mean = rs[0] * (1.f / D_MODEL);
  float var  = rs2[0] * (1.f / D_MODEL) - mean * mean;
  float rstd = rsqrtf(var + 1e-5f);
  const float* sc = proj + (size_t)b * (2 * D_MODEL);
  __bf16* hr = h + (size_t)token * D_MODEL;
  for (int c = threadIdx.x; c < D_MODEL; c += 256) {
    float v = (xr[c] - mean) * rstd;
    v = v * (1.f + sc[c]) + sc[D_MODEL + c];
    hr[c] = (__bf16)v;
  }
}

// ---------------------------------------------------------------------------
// 4) Generic bf16 WMMA GEMM:  C[M,N] = A[M,K] @ W[N,K]^T
//    256 threads = 8 waves, tile 128(M) x 64(N), 2x2 WMMA per wave.
//    Double-buffered LDS tiles streamed by the Tensor Data Mover: the DMA for
//    stage i+1 is issued before computing stage i (wait tensorcnt<=2).
// ---------------------------------------------------------------------------
__global__ __launch_bounds__(256)
void gemm_bf16_wmma(const __bf16* __restrict__ A, const __bf16* __restrict__ W,
                    int M, int N, int K,
                    float* __restrict__ outF, __bf16* __restrict__ outB,
                    const float* __restrict__ residual) {
  const int tid  = threadIdx.x;
  const int w    = tid >> 5;
  const int lane = tid & 31;
  const int ll   = lane & 15;
  const bool hi  = lane >= 16;
  const int wm   = w & 3;          // 0..3 -> 32-row slice
  const int wn   = w >> 2;         // 0..1 -> 32-col slice
  const int m0   = blockIdx.y * 128;
  const int n0   = blockIdx.x * 64;

  __shared__ __align__(16) __bf16 sA[2][128 * 40];  // 32 cols + 8 pad
  __shared__ __align__(16) __bf16 sB[2][64 * 40];

  v8f acc[2][2];
  for (int mi = 0; mi < 2; ++mi)
    for (int ni = 0; ni < 2; ++ni)
      for (int e = 0; e < 8; ++e) acc[mi][ni][e] = 0.f;

  const int nk = K >> 5;

#if USE_TDM
  if (w == 0) {  // uniform per-wave branch; TDM ignores EXEC
    tdm_load_tile(A + (size_t)m0 * K, lds_byte_off(&sA[0][0]), 128, 32, K, 3);
    tdm_load_tile(W + (size_t)n0 * K, lds_byte_off(&sB[0][0]), 64, 32, K, 3);
  }
#endif

  for (int ki = 0; ki < nk; ++ki) {
    const int cur = ki & 1;
#if USE_TDM
    if (w == 0) {
      if (ki + 1 < nk) {
        tdm_load_tile(A + (size_t)m0 * K + (size_t)(ki + 1) * 32,
                      lds_byte_off(&sA[cur ^ 1][0]), 128, 32, K, 3);
        tdm_load_tile(W + (size_t)n0 * K + (size_t)(ki + 1) * 32,
                      lds_byte_off(&sB[cur ^ 1][0]), 64, 32, K, 3);
        __builtin_amdgcn_s_wait_tensorcnt(2);  // stage ki done; ki+1 in flight
      } else {
        __builtin_amdgcn_s_wait_tensorcnt(0);
      }
    }
    __syncthreads();
#else
    { // cooperative stage A: 128 x 32 bf16
      int r = tid >> 1;
      int ch = (tid & 1) * 16;
      const __bf16* src = A + (size_t)(m0 + r) * K + ki * 32 + ch;
      *(v8bf*)&sA[cur][r * 40 + ch]     = *(const v8bf*)(src);
      *(v8bf*)&sA[cur][r * 40 + ch + 8] = *(const v8bf*)(src + 8);
    }
    if (tid < 128) { // cooperative stage W: 64 x 32 bf16
      int r = tid >> 1;
      int ch = (tid & 1) * 16;
      const __bf16* src = W + (size_t)(n0 + r) * K + ki * 32 + ch;
      *(v8bf*)&sB[cur][r * 40 + ch]     = *(const v8bf*)(src);
      *(v8bf*)&sB[cur][r * 40 + ch + 8] = *(const v8bf*)(src + 8);
    }
    __syncthreads();
#endif
    const __bf16* cA = &sA[cur][0];
    const __bf16* cB = &sB[cur][0];

    BF16x16 a[2], bf[2];
    const int aoff = hi ? 8 : 0;     // A split-K: lo {0..7,16..23}, hi {8..15,24..31}
    for (int mi = 0; mi < 2; ++mi) {
      int r = wm * 32 + mi * 16 + ll;
      a[mi].h[0] = *(const v8bf*)&cA[r * 40 + aoff];
      a[mi].h[1] = *(const v8bf*)&cA[r * 40 + 16 + aoff];
    }
    const int boff = hi ? 8 : 0;     // B split-K: lo {0..7,16..23}, hi {8..15,24..31}
    for (int ni = 0; ni < 2; ++ni) {
      int r = wn * 32 + ni * 16 + ll;
      bf[ni].h[0] = *(const v8bf*)&cB[r * 40 + boff];
      bf[ni].h[1] = *(const v8bf*)&cB[r * 40 + boff + 16];
    }
    for (int mi = 0; mi < 2; ++mi)
      for (int ni = 0; ni < 2; ++ni)
        acc[mi][ni] = __builtin_amdgcn_wmma_f32_16x16x32_bf16(
            false, a[mi].v, false, bf[ni].v, (short)0, acc[mi][ni], false, false);
    __syncthreads();   // safe to overwrite buffer 'cur' two stages later
  }

  // epilogue (C layout: vgpr j -> row j(+8 for hi lanes), lane -> col)
  for (int mi = 0; mi < 2; ++mi)
    for (int ni = 0; ni < 2; ++ni)
      for (int jj = 0; jj < 8; ++jj) {
        int row = m0 + wm * 32 + mi * 16 + jj + (hi ? 8 : 0);
        int col = n0 + wn * 32 + ni * 16 + ll;
        float v = acc[mi][ni][jj];
        if (outF) {
          float r = residual ? residual[(size_t)row * N + col] : 0.f;
          outF[(size_t)row * N + col] = v + r;
        } else {
          outB[(size_t)row * N + col] = (__bf16)v;
        }
      }
}

// ---------------------------------------------------------------------------
// 5) RoPE + re-layout [B,T,H,D] -> [B,H,T,D]
// ---------------------------------------------------------------------------
__global__ void rope_arrange_kernel(const __bf16* __restrict__ in,
                                    __bf16* __restrict__ out, int H, int n) {
  int idx = blockIdx.x * blockDim.x + threadIdx.x;
  if (idx >= n) return;
  int d = idx & 127;
  int r = idx >> 7;
  int hh = r % H; r /= H;
  int t = r & 2047;
  int b = r >> 11;
  const __bf16* row = in + ((size_t)((b * T_SEQ + t) * H + hh)) * HD;
  float xv = (float)row[d];
  float xp = (float)row[d ^ 64];
  float rot = (d < 64) ? -xp : xp;
  int i = d & 63;
  float ang = (float)t * __powf(10000.0f, -(float)i * (1.0f / 64.0f));
  float c, s;
  __sincosf(ang, &c, &s);
  out[((size_t)(b * H + hh) * T_SEQ + t) * HD + d] = (__bf16)(xv * c + rot * s);
}

// V: [B,T,KVH,D] -> Vt[B,KVH,D,T]
__global__ void v_transpose_kernel(const __bf16* __restrict__ in,
                                   __bf16* __restrict__ vt, int n) {
  int idx = blockIdx.x * blockDim.x + threadIdx.x;
  if (idx >= n) return;
  int d   = idx & 127;
  int kvh = (idx >> 7) & 3;
  int t   = (idx >> 9) & 2047;
  int b   = idx >> 20;
  vt[((size_t)(b * NKV + kvh) * HD + d) * T_SEQ + t] = in[idx];
}

// ---------------------------------------------------------------------------
// 6) Block-sparse flash attention with WMMA.
//    grid(x = T/128 query blocks, y = B*NH). 8 waves; wave w owns 16 q-rows.
//    K/V tiles for each 32-key chunk are staged ONCE per workgroup into LDS
//    by the TDM (8x traffic reduction vs per-wave global loads).
// ---------------------------------------------------------------------------
__global__ __launch_bounds__(256)
void attn_kernel(const __bf16* __restrict__ qarr,  // [B,NH,T,D]
                 const __bf16* __restrict__ karr,  // [B,NKV,T,D]
                 const __bf16* __restrict__ vt,    // [B,NKV,D,T]
                 __bf16* __restrict__ y) {         // [B,T,NH*D]
  const int qb  = blockIdx.x;
  const int bh  = blockIdx.y;
  const int b   = bh >> 4, hh = bh & 15, kvh = hh >> 2;
  const int tid = threadIdx.x, w = tid >> 5, lane = tid & 31;
  const int ll  = lane & 15;
  const bool hi = lane >= 16;
  const int qt0 = qb * 128;
  const int rbase = qt0 + w * 16;

  __shared__ __align__(16) __bf16 sP[8 * 16 * 40]; // per-wave 16x32 P bounce
  __shared__ __align__(16) __bf16 sK[32 * 136];    // 32 keys x 128 dims (+8 pad)
  __shared__ __align__(16) __bf16 sV[128 * 40];    // 128 dims x 32 times (+8 pad)
  __bf16* myP = sP + w * 16 * 40;

  const __bf16* qbase = qarr + ((size_t)(b * NH  + hh)  * T_SEQ) * HD;
  const __bf16* kbase = karr + ((size_t)(b * NKV + kvh) * T_SEQ) * HD;
  const __bf16* vbase = vt   + ((size_t)(b * NKV + kvh) * HD) * T_SEQ;

  // preload Q fragments (16 rows x 128 dims -> 4 A-frags), reused all chunks
  BF16x16 qa[4];
  {
    const __bf16* qrow = qbase + (size_t)(rbase + ll) * HD;
    const int aoff = hi ? 8 : 0;
    for (int kd = 0; kd < 4; ++kd) {
      qa[kd].h[0] = *(const v8bf*)(qrow + kd * 32 + aoff);
      qa[kd].h[1] = *(const v8bf*)(qrow + kd * 32 + aoff + 16);
    }
  }

  v8f oacc[8];
  for (int i = 0; i < 8; ++i)
    for (int e = 0; e < 8; ++e) oacc[i][e] = 0.f;
  float m_i[8], l_i[8];
  for (int jj = 0; jj < 8; ++jj) { m_i[jj] = -1e30f; l_i[jj] = 0.f; }

  const float sscale = 0.08838834764831845f; // 1/sqrt(128)
  const int send = (qb + 1) * 128;           // block-causal loop bound

  for (int s0 = 0; s0 < send; s0 += 32) {
    // uniform skip: chunk fully outside window with no global column
    if ((s0 + 31 < qt0 - WINDOW) && ((s0 & (GLOBAL_EVERY - 1)) != 0)) continue;

    __syncthreads();   // previous chunk's sK/sV reads complete
#if USE_TDM
    if (w == 0) {
      tdm_load_tile(kbase + (size_t)s0 * HD, lds_byte_off(&sK[0]),
                    32, 128, 128, 5);                 // K rows: 128 el + 16B pad
      tdm_load_tile(vbase + s0, lds_byte_off(&sV[0]),
                    128, 32, T_SEQ, 3);               // V rows: 32 el + 16B pad
      __builtin_amdgcn_s_wait_tensorcnt(0);
    } else if (w == 1 && s0 + 32 < send) {            // hint next K chunk
      __builtin_prefetch((const void*)(kbase + (size_t)(s0 + 32 + lane) * HD), 0, 1);
    }
#else
    { // cooperative K tile: 32 x 128
      int r = tid >> 3, c = (tid & 7) * 16;
      const __bf16* src = kbase + (size_t)(s0 + r) * HD + c;
      *(v8bf*)&sK[r * 136 + c]     = *(const v8bf*)(src);
      *(v8bf*)&sK[r * 136 + c + 8] = *(const v8bf*)(src + 8);
    }
    { // cooperative V tile: 128 x 32
      int r = tid >> 1, c = (tid & 1) * 16;
      const __bf16* src = vbase + (size_t)r * T_SEQ + s0 + c;
      *(v8bf*)&sV[r * 40 + c]     = *(const v8bf*)(src);
      *(v8bf*)&sV[r * 40 + c + 8] = *(const v8bf*)(src + 8);
    }
#endif
    __syncthreads();   // sK/sV visible to all waves

    // S = Q K^T for a 16x32 score tile (2 N-tiles x 4 K-chunks)
    v8f sacc[2];
    for (int j2 = 0; j2 < 2; ++j2) {
      v8f a;
      for (int e = 0; e < 8; ++e) a[e] = 0.f;
      const __bf16* krow = sK + (size_t)(j2 * 16 + ll) * 136;
      for (int kd = 0; kd < 4; ++kd) {
        BF16x16 kb;
        const int boff = kd * 32 + (hi ? 8 : 0);
        kb.h[0] = *(const v8bf*)(krow + boff);
        kb.h[1] = *(const v8bf*)(krow + boff + 16);
        a = __builtin_amdgcn_wmma_f32_16x16x32_bf16(
            false, qa[kd].v, false, kb.v, (short)0, a, false, false);
      }
      sacc[j2] = a;
    }

    // mask + scale, per-row max
    float mrow[8];
    for (int jj = 0; jj < 8; ++jj) mrow[jj] = -1e30f;
    for (int j2 = 0; j2 < 2; ++j2) {
      int s = s0 + j2 * 16 + ll;
      bool glob = (s & (GLOBAL_EVERY - 1)) == 0;
      for (int jj = 0; jj < 8; ++jj) {
        int t = rbase + jj + (hi ? 8 : 0);
        int dd = t > s ? t - s : s - t;
        float sv = sacc[j2][jj] * sscale;
        sv = (glob || dd <= WINDOW) ? sv : -1e30f;
        sacc[j2][jj] = sv;
        mrow[jj] = fmaxf(mrow[jj], sv);
      }
    }
    for (int jj = 0; jj < 8; ++jj) {          // 16-lane row reduction
      float v = mrow[jj];
      v = fmaxf(v, __shfl_xor(v, 1));
      v = fmaxf(v, __shfl_xor(v, 2));
      v = fmaxf(v, __shfl_xor(v, 4));
      v = fmaxf(v, __shfl_xor(v, 8));
      mrow[jj] = v;
    }

    float coef[8], rsum[8];
    for (int jj = 0; jj < 8; ++jj) {
      float mnew = fmaxf(m_i[jj], mrow[jj]);
      coef[jj] = __expf(m_i[jj] - mnew);
      m_i[jj]  = mnew;
      rsum[jj] = 0.f;
    }
    for (int j2 = 0; j2 < 2; ++j2)
      for (int jj = 0; jj < 8; ++jj) {
        float sv = sacc[j2][jj];
        float p  = (sv > -1e29f) ? __expf(sv - m_i[jj]) : 0.f;
        sacc[j2][jj] = p;
        rsum[jj] += p;
      }
    for (int jj = 0; jj < 8; ++jj) {
      float v = rsum[jj];
      v += __shfl_xor(v, 1); v += __shfl_xor(v, 2);
      v += __shfl_xor(v, 4); v += __shfl_xor(v, 8);
      l_i[jj] = l_i[jj] * coef[jj] + v;
    }
    for (int i = 0; i < 8; ++i)
      for (int jj = 0; jj < 8; ++jj) oacc[i][jj] *= coef[jj];

    // P: C-layout -> A-layout via per-wave LDS bounce
    {
      int rofs = hi ? 8 : 0;
      for (int j2 = 0; j2 < 2; ++j2)
        for (int jj = 0; jj < 8; ++jj)
          myP[(rofs + jj) * 40 + j2 * 16 + ll] = (__bf16)sacc[j2][jj];
    }
    BF16x16 pa;
    {
      const int aoff = hi ? 8 : 0;
      pa.h[0] = *(const v8bf*)&myP[ll * 40 + aoff];
      pa.h[1] = *(const v8bf*)&myP[ll * 40 + 16 + aoff];
    }

    // O += P @ V  (sV rows = head dims, contiguous in time)
    for (int i = 0; i < 8; ++i) {
      BF16x16 vb;
      const __bf16* vr = sV + (size_t)(i * 16 + ll) * 40 + (hi ? 8 : 0);
      vb.h[0] = *(const v8bf*)(vr);
      vb.h[1] = *(const v8bf*)(vr + 16);
      oacc[i] = __builtin_amdgcn_wmma_f32_16x16x32_bf16(
          false, pa.v, false, vb.v, (short)0, oacc[i], false, false);
    }
  }

  // normalize and scatter to [B,T,NH*D]
  __bf16* yb = y + (size_t)b * T_SEQ * D_MODEL;
  for (int i = 0; i < 8; ++i)
    for (int jj = 0; jj < 8; ++jj) {
      int t = rbase + jj + (hi ? 8 : 0);
      int col = hh * HD + i * 16 + ll;
      yb[(size_t)t * D_MODEL + col] = (__bf16)(oacc[i][jj] / l_i[jj]);
    }
}

// ---------------------------------------------------------------------------
// Host-side orchestration
// ---------------------------------------------------------------------------
extern "C" void kernel_launch(void* const* d_in, const int* in_sizes, int n_in,
                              void* d_out, int out_size, void* d_ws, size_t ws_size,
                              hipStream_t stream) {
  const float* x     = (const float*)d_in[0];
  const float* t_emb = (const float*)d_in[1];
  const float* W_ada = (const float*)d_in[2];
  const float* b_ada = (const float*)d_in[3];
  const float* Wq    = (const float*)d_in[4];
  const float* Wkd   = (const float*)d_in[5];
  const float* Wku   = (const float*)d_in[6];
  const float* Wvu   = (const float*)d_in[7];
  const float* Wo    = (const float*)d_in[8];

  char* wsp = (char*)d_ws;
  auto alloc = [&](size_t bytes) -> char* {
    char* p = wsp;
    wsp += (bytes + 255) & ~(size_t)255;
    return p;
  };

  const size_t BT = (size_t)BATCH * T_SEQ;          // 4096 tokens
  float*  proj  = (float*) alloc((size_t)BATCH * 2 * D_MODEL * 4);
  __bf16* h     = (__bf16*)alloc(BT * D_MODEL * 2);
  __bf16* wq_b  = (__bf16*)alloc((size_t)D_MODEL * D_MODEL * 2);
  __bf16* wkd_b = (__bf16*)alloc((size_t)LATENT * D_MODEL * 2);
  __bf16* wku_b = (__bf16*)alloc((size_t)LATENT * LATENT * 2);
  __bf16* wvu_b = (__bf16*)alloc((size_t)LATENT * LATENT * 2);
  __bf16* wo_b  = (__bf16*)alloc((size_t)D_MODEL * D_MODEL * 2);
  __bf16* q_pre = (__bf16*)alloc(BT * D_MODEL * 2);
  __bf16* ckv   = (__bf16*)alloc(BT * LATENT * 2);
  __bf16* k_pre = (__bf16*)alloc(BT * LATENT * 2);
  __bf16* v_pre = (__bf16*)alloc(BT * LATENT * 2);
  __bf16* q_arr = (__bf16*)alloc(BT * D_MODEL * 2);
  __bf16* k_arr = (__bf16*)alloc(BT * LATENT * 2);
  __bf16* vt    = (__bf16*)alloc(BT * LATENT * 2);
  __bf16* y     = (__bf16*)alloc(BT * D_MODEL * 2);

  // 1) adaLN projection
  ada_proj_kernel<<<(BATCH * 2 * D_MODEL + 255) / 256, 256, 0, stream>>>(
      t_emb, W_ada, b_ada, proj);

  // 2) weight casts
  {
    int n;
    n = D_MODEL * D_MODEL;
    f32_to_bf16_kernel<<<(n + 255) / 256, 256, 0, stream>>>(Wq,  wq_b,  n);
    f32_to_bf16_kernel<<<(n + 255) / 256, 256, 0, stream>>>(Wo,  wo_b,  n);
    n = LATENT * D_MODEL;
    f32_to_bf16_kernel<<<(n + 255) / 256, 256, 0, stream>>>(Wkd, wkd_b, n);
    n = LATENT * LATENT;
    f32_to_bf16_kernel<<<(n + 255) / 256, 256, 0, stream>>>(Wku, wku_b, n);
    f32_to_bf16_kernel<<<(n + 255) / 256, 256, 0, stream>>>(Wvu, wvu_b, n);
  }

  // 3) LN + modulation
  ln_mod_kernel<<<(int)BT, 256, 0, stream>>>(x, proj, h);

  // 4) projection GEMMs (C = A @ W^T)
  gemm_bf16_wmma<<<dim3(D_MODEL / 64, (int)BT / 128), 256, 0, stream>>>(
      h, wq_b, (int)BT, D_MODEL, D_MODEL, nullptr, q_pre, nullptr);
  gemm_bf16_wmma<<<dim3(LATENT / 64, (int)BT / 128), 256, 0, stream>>>(
      h, wkd_b, (int)BT, LATENT, D_MODEL, nullptr, ckv, nullptr);
  gemm_bf16_wmma<<<dim3(LATENT / 64, (int)BT / 128), 256, 0, stream>>>(
      ckv, wku_b, (int)BT, LATENT, LATENT, nullptr, k_pre, nullptr);
  gemm_bf16_wmma<<<dim3(LATENT / 64, (int)BT / 128), 256, 0, stream>>>(
      ckv, wvu_b, (int)BT, LATENT, LATENT, nullptr, v_pre, nullptr);

  // 5) RoPE + layout
  {
    int nq = BATCH * T_SEQ * NH * HD;
    rope_arrange_kernel<<<(nq + 255) / 256, 256, 0, stream>>>(q_pre, q_arr, NH, nq);
    int nk = BATCH * T_SEQ * NKV * HD;
    rope_arrange_kernel<<<(nk + 255) / 256, 256, 0, stream>>>(k_pre, k_arr, NKV, nk);
    v_transpose_kernel<<<(nk + 255) / 256, 256, 0, stream>>>(v_pre, vt, nk);
  }

  // 6) block-sparse flash attention
  attn_kernel<<<dim3(T_SEQ / 128, BATCH * NH), 256, 0, stream>>>(
      q_arr, k_arr, vt, y);

  // 7) output projection + residual (fp32 out)
  gemm_bf16_wmma<<<dim3(D_MODEL / 64, (int)BT / 128), 256, 0, stream>>>(
      y, wo_b, (int)BT, D_MODEL, D_MODEL, (float*)d_out, nullptr, x);
}
